// StrangeAttractor_45183055954393
// MI455X (gfx1250) — compile-verified, matrix operand order checked by reading, not run on hardware
//
#include <hip/hip_runtime.h>

typedef __attribute__((ext_vector_type(16))) _Float16 v16h;
typedef __attribute__((ext_vector_type(8)))  float    v8f;
typedef __attribute__((ext_vector_type(4)))  float    f4;

#define B_ROWS 524288
#define E_DIM  128

__global__ __launch_bounds__(256, 4) void StrangeAttractor_kernel(
    const float* __restrict__ X,     // [B, E] expert_activations
    const float* __restrict__ C,     // [E, E] attractor_centers
    const float* __restrict__ R,     // [E]    attraction_radii
    float* __restrict__ Out,         // [B, E] attracted
    float* __restrict__ OutIdx)      // [B]    idx (as float)
{
    // f16 B-operand tiles in WMMA lane layout, [ktile][ntile][lane][16 halves] (~33KB LDS)
    __shared__ alignas(32) _Float16 sB[4 * 8 * 32 * 16];
    __shared__ float sCsq[E_DIM];

    const int tid  = threadIdx.x;
    const int lane = tid & 31;
    const int w    = tid >> 5;       // wave in block (0..7)
    const int hi   = lane >> 4;      // half-wave group
    const int nl   = lane & 15;

    // ---- one-time per block: ||c||^2 and f16 B-operand pack (C is L2-hot, 64KB) ----
    if (tid < E_DIM) {
        float s = 0.f;
        for (int k = 0; k < E_DIM; ++k) { float v = C[tid * E_DIM + k]; s += v * v; }
        sCsq[tid] = s;
    }
    // sB[k][n][l][h] = c[16n + (l&15)][32k + Klocal(h, l>>4)] as f16
    for (int i = tid; i < 4 * 8 * 32 * 16; i += 256) {
        int h   = i & 15;
        int l   = (i >> 4) & 31;
        int n   = (i >> 9) & 7;
        int k   = i >> 12;
        int lhi = l >> 4;
        int K   = 32 * k + h + (h < 8 ? 0 : 8) + 8 * lhi;
        int N   = 16 * n + (l & 15);
        sB[i] = (_Float16)C[N * E_DIM + K];
    }
    __syncthreads();

    const int ntile_cnt = B_ROWS / 128;
    for (int tile = blockIdx.x; tile < ntile_cnt; tile += gridDim.x) {
        const int rowbase = tile * 128 + w * 16;
        const float* xrow = X + (size_t)(rowbase + nl) * E_DIM;

        // Opaque zero, re-materialized every iteration: prevents LICM from hoisting
        // the 32 B-operand LDS loads out of the tile loop (would spill to scratch).
        int zofs;
        asm volatile("s_mov_b32 %0, 0" : "=s"(zofs));
        const _Float16* sBt = sB + zofs;

        // Prefetch next tile's x slab (each lane covers 2 cachelines of its row half).
        const int tnext = tile + gridDim.x;
        if (tnext < ntile_cnt) {
            const float* xn = X + (size_t)(tnext * 128 + w * 16 + nl) * E_DIM + hi * 64;
            __builtin_prefetch(xn, 0, 3);
            __builtin_prefetch(xn + 32, 0, 3);
        }

        // ---- A operand: 16x128 f16, plus exact fp32 ||x||^2 per row ----
        v16h a[4];
        float xsq = 0.f;
        #pragma unroll
        for (int k = 0; k < 4; ++k) {
            const int o0 = 32 * k + 8 * hi;
            f4 q0 = *(const f4*)(xrow + o0);
            f4 q1 = *(const f4*)(xrow + o0 + 4);
            f4 q2 = *(const f4*)(xrow + o0 + 16);
            f4 q3 = *(const f4*)(xrow + o0 + 20);
            v16h t;
            t[0]  = (_Float16)q0.x; t[1]  = (_Float16)q0.y; t[2]  = (_Float16)q0.z; t[3]  = (_Float16)q0.w;
            t[4]  = (_Float16)q1.x; t[5]  = (_Float16)q1.y; t[6]  = (_Float16)q1.z; t[7]  = (_Float16)q1.w;
            t[8]  = (_Float16)q2.x; t[9]  = (_Float16)q2.y; t[10] = (_Float16)q2.z; t[11] = (_Float16)q2.w;
            t[12] = (_Float16)q3.x; t[13] = (_Float16)q3.y; t[14] = (_Float16)q3.z; t[15] = (_Float16)q3.w;
            a[k] = t;
            xsq += q0.x*q0.x + q0.y*q0.y + q0.z*q0.z + q0.w*q0.w
                 + q1.x*q1.x + q1.y*q1.y + q1.z*q1.z + q1.w*q1.w
                 + q2.x*q2.x + q2.y*q2.y + q2.z*q2.z + q2.w*q2.w
                 + q3.x*q3.x + q3.y*q3.y + q3.z*q3.z + q3.w*q3.w;
        }
        xsq += __shfl_xor(xsq, 16);          // combine the two K-halves of row (lane&15)

        float xs[8];
        #pragma unroll
        for (int r = 0; r < 8; ++r)          // ||x||^2 for row M = r + 8*hi
            xs[r] = __shfl(xsq, r + 8 * hi);

        // ---- GEMM: 32 WMMAs, 8 independent accumulator chains, B double-buffered ----
        v8f acc[8];
        #pragma unroll
        for (int n = 0; n < 8; ++n) acc[n] = (v8f){};

        v16h bc = *(const v16h*)(sBt + (lane << 4));   // j = 0
        #pragma unroll
        for (int j = 0; j < 32; ++j) {
            v16h bn = bc;
            if (j < 31)                                 // prefetch B for j+1
                bn = *(const v16h*)(sBt + ((((j + 1) * 32) + lane) << 4));
            acc[j & 7] = __builtin_amdgcn_wmma_f32_16x16x32_f16(
                             false, a[j >> 3], false, bc, (short)0, acc[j & 7], false, false);
            bc = bn;
        }

        // ---- d2 and per-lane running argmin over column tiles ----
        float minv[8]; int mini[8];
        #pragma unroll
        for (int r = 0; r < 8; ++r) { minv[r] = 3.4e38f; mini[r] = 0; }

        #pragma unroll
        for (int n = 0; n < 8; ++n) {
            const int col  = 16 * n + nl;
            const float cs = sCsq[col];
            #pragma unroll
            for (int r = 0; r < 8; ++r) {
                const float d2 = xs[r] + cs - 2.0f * acc[n][r];
                if (d2 < minv[r]) { minv[r] = d2; mini[r] = col; }
            }
        }

        // ---- argmin across the 16-lane N-groups (wave32-safe xor masks) ----
        #pragma unroll
        for (int m = 1; m < 16; m <<= 1) {
            #pragma unroll
            for (int r = 0; r < 8; ++r) {
                const float ov = __shfl_xor(minv[r], m);
                const int   oi = __shfl_xor(mini[r], m);
                if (ov < minv[r] || (ov == minv[r] && oi < mini[r])) {
                    minv[r] = ov; mini[r] = oi;
                }
            }
        }

        float sval[8];
        #pragma unroll
        for (int r = 0; r < 8; ++r) {
            const float dist = sqrtf(fmaxf(minv[r], 0.f));
            const float rad  = R[mini[r]];
            sval[r] = 0.1f * expf(-dist / (rad + 1e-8f));
        }

        // ---- blended output: each row written as 32 lanes x float4; c[idx] from L2 ----
        #pragma unroll
        for (int ro = 0; ro < 16; ++ro) {
            const int r = ro & 7, g = ro >> 3;
            const float sv = __shfl(sval[r], g * 16);
            const int   ix = __shfl(mini[r], g * 16);
            const size_t row = (size_t)(rowbase + ro);
            const f4 xv = *(const f4*)(X + row * E_DIM + 4 * lane);
            const f4 cv = *(const f4*)(C + (size_t)ix * E_DIM + 4 * lane);
            *(f4*)(Out + row * E_DIM + 4 * lane) = xv * (1.0f - sv) + cv * sv;
            if (lane == 0) OutIdx[row] = (float)ix;
        }
    }
}

extern "C" void kernel_launch(void* const* d_in, const int* in_sizes, int n_in,
                              void* d_out, int out_size, void* d_ws, size_t ws_size,
                              hipStream_t stream) {
    const float* X = (const float*)d_in[0];   // expert_activations [B, E]
    const float* C = (const float*)d_in[1];   // attractor_centers  [E, E]
    const float* R = (const float*)d_in[2];   // attraction_radii   [E]
    float* Out    = (float*)d_out;                          // attracted [B, E]
    float* OutIdx = (float*)d_out + (size_t)B_ROWS * E_DIM; // idx [B]

    StrangeAttractor_kernel<<<2048, 256, 0, stream>>>(X, C, R, Out, OutIdx);
}
